// RGCNClassifier_88648124990446
// MI455X (gfx1250) — matrix operand order repaired and checked by reading, not compiled, not run on hardware
//
#include <hip/hip_runtime.h>

#define NN 50000
#define NE 1250000
#define FD 64
#define NG 512
#define NC 2

typedef __attribute__((ext_vector_type(16))) __bf16 v16bf;
typedef __attribute__((ext_vector_type(8)))  float  v8f;

__device__ __forceinline__ unsigned f32_bf16(float f) {
  union { float f; unsigned u; } c; c.f = f;
  unsigned u = c.u;
  u += 0x7FFFu + ((u >> 16) & 1u);   // round-to-nearest-even
  return u >> 16;
}
__device__ __forceinline__ unsigned pack_bf16(float lo, float hi) {
  return (f32_bf16(hi) << 16) | f32_bf16(lo);
}

__global__ void zero_f32(float* __restrict__ p, int n) {
  int i = blockIdx.x * blockDim.x + threadIdx.x;
  int stride = gridDim.x * blockDim.x;
  for (; i < n; i += stride) p[i] = 0.0f;
}

// h[n,f] = (x[n]==0) ? 0 : emb[x[n],f]   (padding_idx = 0)
__global__ void embed_kernel(const int* __restrict__ x, const float* __restrict__ emb,
                             float* __restrict__ h) {
  int t = blockIdx.x * blockDim.x + threadIdx.x;
  if (t >= NN * FD) return;
  int n = t >> 6, f = t & 63;
  int tok = x[n];
  h[t] = (tok == 0) ? 0.0f : emb[tok * FD + f];
}

// Scatter-add per edge: agg[r][dst][f] += h[src][f]; cnt[r][dst] += 1
__global__ void aggregate_kernel(const float* __restrict__ h, const int* __restrict__ src,
                                 const int* __restrict__ dst, const int* __restrict__ et,
                                 float* __restrict__ agg, float* __restrict__ cnt) {
  long long t = (long long)blockIdx.x * blockDim.x + threadIdx.x;
  int e = (int)(t >> 6);
  int f = (int)(t & 63);
  if (e >= NE) return;
  int s = src[e], d = dst[e], r = et[e];
  atomicAdd(&agg[(r * NN + d) * FD + f], h[s * FD + f]);
  if (f == 0) atomicAdd(&cnt[r * NN + d], 1.0f);
}

// out = ReLU( h@Wroot + b + sum_r (agg_r / max(cnt_r,1)) @ W_r )
// One wave per 16-row node tile; fused K=256 GEMM (4 sources x 2 k-chunks),
// 4 n-tiles of 16 cols, bf16 WMMA with f32 accumulation.
// LDS holds fragments in register order so loads are ds_load_b128 pairs;
// staging uses float4 global loads + ds_store_b64 quads (aligned 4-groups of k
// stay contiguous under the A-fragment mapping).
__global__ void __launch_bounds__(256)
rgcn_transform(const float* __restrict__ h, const float* __restrict__ agg,
               const float* __restrict__ cnt, const float* __restrict__ w_root,
               const float* __restrict__ w_rel, const float* __restrict__ bias,
               float* __restrict__ out) {
  // [s][chunk][nt][lane][e] -> B fragments, 4*2*4*32*16 ushort = 32 KB
  __shared__ __attribute__((aligned(32))) unsigned short wfrag[4 * 2 * 4 * 32 * 16];
  // per-wave A tile fragments [chunk][lane][e], 1024 ushort = 2 KB each
  __shared__ __attribute__((aligned(32))) unsigned short tfrag[8][2 * 32 * 16];
  __shared__ float blds[64];

  int tid = threadIdx.x;
  // Stage weights (root + 3 rel) into B-fragment layout:
  //   B 32x16 (16-bit): lane = (n&15) + 16*(kl>>4), e = kl&15, kl = k&31
  for (int idx = tid; idx < 4 * 64 * 64; idx += 256) {
    int s = idx >> 12, rem = idx & 4095;
    int k = rem >> 6, n = rem & 63;
    float w = (s == 0) ? w_root[rem] : w_rel[(s - 1) * 4096 + rem];
    int chunk = k >> 5, kl = k & 31;
    int laneB = (n & 15) + 16 * (kl >> 4);
    int e = kl & 15;
    int nt = n >> 4;
    wfrag[(((s * 2 + chunk) * 4 + nt) * 32 + laneB) * 16 + e] =
        (unsigned short)f32_bf16(w);
  }
  if (tid < 64) blds[tid] = bias[tid];
  __syncthreads();

  int lane = tid & 31, wave = tid >> 5;
  int lh = lane >> 4;          // lane half (0/1)
  int mn = lane & 15;          // col within C tile / row within A
  int rowBase = (blockIdx.x * 8 + wave) * 16;

  v8f acc[4] = {};
  unsigned short* mytile = tfrag[wave];

  for (int s = 0; s < 4; ++s) {
    // Stage X_s (16x64): 8 float4 loads per lane, ds_store_b64 quads into
    // A-fragment layout (half=kl>>4, e=kl&15, lane=m+16*half)
    #pragma unroll
    for (int i = 0; i < 8; ++i) {
      int flat4 = i * 32 + lane;       // float4 index within 16x16 quad grid
      int r16 = flat4 >> 4;            // row 0..15
      int c4 = (flat4 & 15) * 4;       // col 0,4,...,60
      int node = rowBase + r16;
      node = node < NN ? node : NN - 1;   // clamp (stores guarded later)
      float4 v = *(const float4*)&h[node * FD + c4];
      if (s != 0) {
        int ni = (s - 1) * NN + node;
        float4 a = *(const float4*)&agg[ni * FD + c4];
        float inv = __builtin_amdgcn_rcpf(fmaxf(cnt[ni], 1.0f));
        v.x = a.x * inv; v.y = a.y * inv; v.z = a.z * inv; v.w = a.w * inv;
      }
      int chunk = c4 >> 5, kl = c4 & 31;
      int halfA = (kl >> 4) & 1;
      int e = kl & 15;
      int laneA = r16 + 16 * halfA;
      uint2 pk;
      pk.x = pack_bf16(v.x, v.y);
      pk.y = pack_bf16(v.z, v.w);
      *(uint2*)&mytile[(chunk * 32 + laneA) * 16 + e] = pk;
    }
    __syncthreads();

    // A fragments: one 32B vector load per chunk (ds_load_b128 x2 each)
    v16bf a0 = *(const v16bf*)&mytile[(0 * 32 + lane) * 16];
    v16bf a1 = *(const v16bf*)&mytile[(1 * 32 + lane) * 16];

    #pragma unroll
    for (int nt = 0; nt < 4; ++nt) {
      v16bf b0 = *(const v16bf*)&wfrag[(((s * 2 + 0) * 4 + nt) * 32 + lane) * 16];
      v16bf b1 = *(const v16bf*)&wfrag[(((s * 2 + 1) * 4 + nt) * 32 + lane) * 16];
      acc[nt] = __builtin_amdgcn_wmma_f32_16x16x32_bf16(
          false, a0, false, b0, (short)0, acc[nt], false, false);
      acc[nt] = __builtin_amdgcn_wmma_f32_16x16x32_bf16(
          false, a1, false, b1, (short)0, acc[nt], false, false);
    }
    __syncthreads();   // tile safe to overwrite next iteration
  }

  // Epilogue: C/D layout -> VGPR j holds row (j + 8*half), col = mn + 16*nt
  #pragma unroll
  for (int nt = 0; nt < 4; ++nt) {
    int ncol = nt * 16 + mn;
    float bv = blds[ncol];
    #pragma unroll
    for (int j = 0; j < 8; ++j) {
      int row = rowBase + j + 8 * lh;
      if (row < NN) out[row * FD + ncol] = fmaxf(acc[nt][j] + bv, 0.0f);
    }
  }
}

// Global mean pool: psum[batch[n]][f] += h[n][f]; pcnt[batch[n]] += 1
__global__ void pool_kernel(const float* __restrict__ h, const int* __restrict__ batch,
                            float* __restrict__ psum, float* __restrict__ pcnt) {
  int t = blockIdx.x * blockDim.x + threadIdx.x;
  if (t >= NN * FD) return;
  int n = t >> 6, f = t & 63;
  int g = batch[n];
  atomicAdd(&psum[g * FD + f], h[t]);
  if (f == 0) atomicAdd(&pcnt[g], 1.0f);
}

// out[g,c] = lin_b[c] + (psum[g]/max(pcnt[g],1)) . lin_w[:,c]
__global__ void head_kernel(const float* __restrict__ psum, const float* __restrict__ pcnt,
                            const float* __restrict__ lw, const float* __restrict__ lb,
                            float* __restrict__ out) {
  int t = blockIdx.x * blockDim.x + threadIdx.x;
  if (t >= NG * NC) return;
  int g = t >> 1, c = t & 1;
  float inv = 1.0f / fmaxf(pcnt[g], 1.0f);
  float acc = lb[c];
  #pragma unroll 8
  for (int f = 0; f < FD; ++f) acc += psum[g * FD + f] * inv * lw[f * NC + c];
  out[t] = acc;
}

extern "C" void kernel_launch(void* const* d_in, const int* in_sizes, int n_in,
                              void* d_out, int out_size, void* d_ws, size_t ws_size,
                              hipStream_t stream) {
  const int*   x       = (const int*)d_in[0];
  const int*   ei      = (const int*)d_in[1];   // [2, NE]: row0 = src, row1 = dst
  const int*   et      = (const int*)d_in[2];
  const int*   batch   = (const int*)d_in[3];
  const float* emb     = (const float*)d_in[4];
  const float* w1_rel  = (const float*)d_in[5];
  const float* w1_root = (const float*)d_in[6];
  const float* b1      = (const float*)d_in[7];
  const float* w2_rel  = (const float*)d_in[8];
  const float* w2_root = (const float*)d_in[9];
  const float* b2      = (const float*)d_in[10];
  const float* lw      = (const float*)d_in[11];
  const float* lb      = (const float*)d_in[12];
  float* outp = (float*)d_out;

  // Workspace layout (f32): ~65 MB total
  float* ws   = (float*)d_ws;
  float* hA   = ws;                       // NN*FD
  float* hB   = hA + NN * FD;             // NN*FD
  float* agg  = hB + NN * FD;             // 3*NN*FD
  float* cnt  = agg + 3 * NN * FD;        // 3*NN   (contiguous after agg)
  float* psum = cnt + 3 * NN;             // NG*FD
  float* pcnt = psum + NG * FD;           // NG     (contiguous after psum)

  const int* srcp = ei;
  const int* dstp = ei + NE;

  const int nf = NN * FD;
  const int zAgg = 3 * NN * FD + 3 * NN;
  const int aggBlocks = (int)(((long long)NE * 64 + 255) / 256);
  const int xfBlocks = (NN + 127) / 128;

  embed_kernel<<<(nf + 255) / 256, 256, 0, stream>>>(x, emb, hA);

  // Layer 1
  zero_f32<<<2048, 256, 0, stream>>>(agg, zAgg);
  aggregate_kernel<<<aggBlocks, 256, 0, stream>>>(hA, srcp, dstp, et, agg, cnt);
  rgcn_transform<<<xfBlocks, 256, 0, stream>>>(hA, agg, cnt, w1_root, w1_rel, b1, hB);

  // Layer 2
  zero_f32<<<2048, 256, 0, stream>>>(agg, zAgg);
  aggregate_kernel<<<aggBlocks, 256, 0, stream>>>(hB, srcp, dstp, et, agg, cnt);
  rgcn_transform<<<xfBlocks, 256, 0, stream>>>(hB, agg, cnt, w2_root, w2_rel, b2, hA);

  // Pool + head
  zero_f32<<<128, 256, 0, stream>>>(psum, NG * FD + NG);
  pool_kernel<<<(nf + 255) / 256, 256, 0, stream>>>(hA, batch, psum, pcnt);
  head_kernel<<<(NG * NC + 255) / 256, 256, 0, stream>>>(psum, pcnt, lw, lb, outp);
}
